// StructureGuidedContrastiveLoss_17910013624523
// MI455X (gfx1250) — matrix operand order repaired and checked by reading, not compiled, not run on hardware
//
#include <hip/hip_runtime.h>
#include <hip/hip_bf16.h>

#define NN 8192
#define DD 256
#define INV_T 5.0f   // 1/TEMPERATURE, folded into zv normalization

typedef __attribute__((ext_vector_type(16))) __bf16 v16bf;
typedef __attribute__((ext_vector_type(8)))  __bf16 v8bf;
typedef __attribute__((ext_vector_type(8)))  float  v8f;

// ---------------------------------------------------------------------------
// Kernel 1: row L2-normalize zv (scaled by 1/T) and zf, emit bf16 copies.
// One wave (32 lanes) per row; 8 waves per block.
// ---------------------------------------------------------------------------
__global__ __launch_bounds__(256) void normalize_bf16_kernel(
    const float* __restrict__ zv, const float* __restrict__ zf,
    __bf16* __restrict__ zvb, __bf16* __restrict__ zfb)
{
    const int gwave = (blockIdx.x * blockDim.x + threadIdx.x) >> 5;
    const int lane  = threadIdx.x & 31;

    const float* src;
    __bf16* dst;
    float extra;
    if (gwave < NN) {
        src = zv + (size_t)gwave * DD;
        dst = zvb + (size_t)gwave * DD;
        extra = INV_T;
    } else {
        const int r = gwave - NN;
        src = zf + (size_t)r * DD;
        dst = zfb + (size_t)r * DD;
        extra = 1.0f;
    }

    float ss = 0.0f;
    #pragma unroll
    for (int j = lane; j < DD; j += 32) { float v = src[j]; ss += v * v; }
    #pragma unroll
    for (int off = 16; off >= 1; off >>= 1) ss += __shfl_xor(ss, off, 32);

    const float inv = extra / fmaxf(sqrtf(ss), 1e-12f);
    #pragma unroll
    for (int j = lane; j < DD; j += 32) dst[j] = (__bf16)(src[j] * inv);
}

// ---------------------------------------------------------------------------
// Kernel 2: fused sim-tile WMMA + mask + online partial reductions.
// Block = 16 output rows (8 waves). Wave w processes column tiles w, w+8, ...
// Never materializes the 8192x8192 sim matrix. Gf is streamed non-temporally
// (read exactly once; keep the 192MB L2 for the reused bf16 embeddings).
// ---------------------------------------------------------------------------
__global__ __launch_bounds__(256) void simloss_kernel(
    const __bf16* __restrict__ zvb, const __bf16* __restrict__ zfb,
    const float*  __restrict__ Gf,  float* __restrict__ partials)
{
    __shared__ float red[8][16][3];
    __shared__ float red2[16];

    const int tid    = threadIdx.x;
    const int wave   = tid >> 5;
    const int lane   = tid & 31;
    const int hl     = lane & 15;       // column index within tile (B/C/D), row for A
    const bool hiH   = lane >= 16;
    const int rbase  = blockIdx.x * 16;

    // ---- A operand: 16 rows x 256 K of bf16, resident in registers ----
    // Lane layout: lanes 0-15 hold K {32k+0..15};
    // lanes 16-31 hold K {32k+16..31}.
    const __bf16* aRow = zvb + (size_t)(rbase + hl) * DD + (hiH ? 16 : 0);
    v16bf a[8];
    #pragma unroll
    for (int k = 0; k < 8; ++k) {
        v8bf lo  = *(const v8bf*)(aRow + 32 * k);
        v8bf hi8 = *(const v8bf*)(aRow + 32 * k + 8);
        v16bf t;
        #pragma unroll
        for (int i = 0; i < 8; ++i) { t[i] = lo[i]; t[8 + i] = hi8[i]; }
        a[k] = t;
    }

    float sumexp[8], psum[8], pcnt[8];
    #pragma unroll
    for (int r = 0; r < 8; ++r) { sumexp[r] = 0.f; psum[r] = 0.f; pcnt[r] = 0.f; }

    const int mb = rbase + (hiH ? 8 : 0);   // first output row held by this lane

    for (int ct = wave; ct < NN / 16; ct += 8) {
        const int cbase = ct * 16;
        // B operand: 16 contiguous bf16 (32B) per lane per K-step.
        const __bf16* bRow = zfb + (size_t)(cbase + hl) * DD + (hiH ? 16 : 0);

        v8f acc = {};
        #pragma unroll
        for (int k = 0; k < 8; ++k) {
            v16bf b = *(const v16bf*)(bRow + 32 * k);
            acc = __builtin_amdgcn_wmma_f32_16x16x32_bf16(
                false, a[k], false, b, (short)0, acc, false, false);
        }

        const int col = cbase + hl;
        const float* gptr = Gf + (size_t)mb * NN + col;
        #pragma unroll
        for (int r = 0; r < 8; ++r) {
            const float g = __builtin_nontemporal_load(gptr + (size_t)r * NN);
            const float s = acc[r];
            const bool pos = (g > 0.0f) || ((mb + r) == col);
            sumexp[r] += pos ? 1.0f : __expf(s);   // masked slot stays as logit 0
            psum[r]   += pos ? s : 0.0f;
            pcnt[r]   += pos ? 1.0f : 0.0f;
        }
    }

    // Reduce across the 16 lanes of each half-wave (columns of the tile).
    #pragma unroll
    for (int off = 1; off <= 8; off <<= 1) {
        #pragma unroll
        for (int r = 0; r < 8; ++r) {
            sumexp[r] += __shfl_xor(sumexp[r], off, 32);
            psum[r]   += __shfl_xor(psum[r],   off, 32);
            pcnt[r]   += __shfl_xor(pcnt[r],   off, 32);
        }
    }

    if (hl == 0) {  // lanes 0 (rows 0-7) and 16 (rows 8-15)
        const int ro = hiH ? 8 : 0;
        #pragma unroll
        for (int r = 0; r < 8; ++r) {
            red[wave][ro + r][0] = sumexp[r];
            red[wave][ro + r][1] = psum[r];
            red[wave][ro + r][2] = pcnt[r];
        }
    }
    __syncthreads();

    if (tid < 16) {
        float se = 0.f, ps = 0.f, pc = 0.f;
        #pragma unroll
        for (int w = 0; w < 8; ++w) {
            se += red[w][tid][0];
            ps += red[w][tid][1];
            pc += red[w][tid][2];
        }
        const float pos_score = ps / (pc + 1e-8f);
        // lse over [pos_score, masked row]; |s|<=5 so no max-shift needed.
        red2[tid] = __logf(se + __expf(pos_score)) - pos_score;
    }
    __syncthreads();

    if (tid == 0) {
        float s = 0.f;
        #pragma unroll
        for (int i = 0; i < 16; ++i) s += red2[i];
        partials[blockIdx.x] = s;
    }
}

// ---------------------------------------------------------------------------
// Kernel 3: deterministic final reduction of 512 block partials -> loss / N.
// ---------------------------------------------------------------------------
__global__ __launch_bounds__(256) void finalize_kernel(
    const float* __restrict__ partials, float* __restrict__ out)
{
    __shared__ float sh[256];
    const int tid = threadIdx.x;
    sh[tid] = partials[tid] + partials[tid + 256];
    __syncthreads();
    #pragma unroll
    for (int s = 128; s > 0; s >>= 1) {
        if (tid < s) sh[tid] += sh[tid + s];
        __syncthreads();
    }
    if (tid == 0) out[0] = sh[0] / (float)NN;
}

// ---------------------------------------------------------------------------
extern "C" void kernel_launch(void* const* d_in, const int* in_sizes, int n_in,
                              void* d_out, int out_size, void* d_ws, size_t ws_size,
                              hipStream_t stream)
{
    const float* zv = (const float*)d_in[0];   // z_view  [N, D] f32
    const float* zf = (const float*)d_in[1];   // z_fused [N, D] f32
    const float* Gf = (const float*)d_in[2];   // Gf_dense [N, N] f32

    char* ws = (char*)d_ws;
    __bf16* zvb      = (__bf16*)ws;                                 // 4 MB
    __bf16* zfb      = (__bf16*)(ws + (size_t)NN * DD * 2);         // 4 MB
    float*  partials = (float*)(ws + (size_t)NN * DD * 4);          // 2 KB

    // 2*N rows, one wave each, 8 waves per 256-thread block.
    normalize_bf16_kernel<<<(2 * NN) / 8, 256, 0, stream>>>(zv, zf, zvb, zfb);

    // 512 row-tiles of 16 rows.
    simloss_kernel<<<NN / 16, 256, 0, stream>>>(zvb, zfb, Gf, partials);

    finalize_kernel<<<1, 256, 0, stream>>>(partials, (float*)d_out);
}
